// MultiHeadAttention_1219770712401
// MI455X (gfx1250) — matrix operand order repaired and checked by reading, not compiled, not run on hardware
//
#include <hip/hip_runtime.h>
#include <hip/hip_bf16.h>

typedef __bf16 bf16_t;
typedef __attribute__((ext_vector_type(16))) __bf16 v16bf;
typedef __attribute__((ext_vector_type(8)))  float  v8f;
typedef __attribute__((ext_vector_type(4)))  unsigned int u32x4;
typedef __attribute__((ext_vector_type(8)))  unsigned int u32x8;

#define B_SZ   4
#define T_SZ   2048
#define D_SZ   1024
#define H_SZ   16
#define HD_SZ  64
#define M_TOT  (B_SZ * T_SZ)   // 8192

// ---------------------------------------------------------------------------
// Fragment loaders: A and B both use contiguous k order, element s of lane l -> k = 16*(l>>4) + s
// ---------------------------------------------------------------------------

// A fragment: 16x32 bf16 (MxK) tile at `base`, row stride ld (elements).
// lane<16: row=lane,     K = {0..15}
// lane>=16: row=lane-16, K = {16..31}
__device__ __forceinline__ v16bf load_a_frag(const bf16_t* base, int ld) {
  int l = threadIdx.x & 31;
  int m = l & 15, h = l >> 4;
  const bf16_t* p0 = base + (size_t)m * ld + h * 16;
  const bf16_t* p1 = base + (size_t)m * ld + h * 16 + 8;
  v16bf f;
#pragma unroll
  for (int j = 0; j < 8; ++j) f[j] = p0[j];
#pragma unroll
  for (int j = 0; j < 8; ++j) f[8 + j] = p1[j];
  return f;
}

// B fragment: 32x16 bf16 (KxN) where column n of B is a contiguous row in
// memory (true for nn.Linear weights W[n][k] and for K^T in attention).
__device__ __forceinline__ v16bf load_b_rowmajor(const bf16_t* base, int ld) {
  int l = threadIdx.x & 31;
  int n = l & 15, h = l >> 4;
  const bf16_t* p = base + (size_t)n * ld + h * 16;
  v16bf f;
#pragma unroll
  for (int j = 0; j < 16; ++j) f[j] = p[j];
  return f;
}

// B fragment: 32x16 bf16 (KxN) stored row-major as B[k][n], row stride ld
// (used for V in the PV product; strided gather along K).
__device__ __forceinline__ v16bf load_b_colstride(const bf16_t* base, int ld) {
  int l = threadIdx.x & 31;
  int n = l & 15, h = l >> 4;
  const bf16_t* p = base + (size_t)(h * 16) * ld + n;
  v16bf f;
#pragma unroll
  for (int j = 0; j < 16; ++j) f[j] = p[(size_t)j * ld];
  return f;
}

// ---------------------------------------------------------------------------
// Tensor Data Mover: async 2D tile (tile_d0 contiguous elems x tile_d1 rows,
// bf16) from global to LDS.  D# per CDNA5 ISA.
// Tracked by TENSORcnt; pair with __builtin_amdgcn_s_wait_tensorcnt.
// ---------------------------------------------------------------------------
__device__ __forceinline__ void tdm_load_2d_bf16(unsigned lds_addr,
                                                 const void* gptr,
                                                 unsigned tensor_d0,
                                                 unsigned tensor_d1,
                                                 unsigned tile_d0,
                                                 unsigned tile_d1,
                                                 unsigned stride0) {
  unsigned long long ga = (unsigned long long)(uintptr_t)gptr;
  u32x4 g0;
  g0[0] = 1u;                                  // count=1 (valid user D#)
  g0[1] = lds_addr;                            // lds_addr[31:0] -> bits 63:32
  g0[2] = (unsigned)(ga & 0xffffffffull);      // global_addr[31:0]
  g0[3] = (unsigned)((ga >> 32) & 0x1ffffffull) | (2u << 30);  // addr[56:32], type=2
  u32x8 g1;
  g1[0] = 1u << 16;                            // workgroup_mask=0, data_size=1 (2B)
  g1[1] = tensor_d0 << 16;                     // tensor_dim0[15:0] @ [63:48]
  g1[2] = (tensor_d0 >> 16) | (tensor_d1 << 16);   // dim0[31:16] | dim1[15:0]
  g1[3] = (tensor_d1 >> 16) | (tile_d0 << 16);     // dim1[31:16] | tile_dim0
  g1[4] = tile_d1 & 0xffffu;                   // tile_dim1 (tile_dim2 = 0)
  g1[5] = stride0;                             // tensor_dim0_stride[31:0]
  g1[6] = 0u;
  g1[7] = 0u;
  asm volatile("tensor_load_to_lds %0, %1" :: "s"(g0), "s"(g1) : "memory");
}

// ---------------------------------------------------------------------------
// f32 -> bf16 conversion
// ---------------------------------------------------------------------------
__global__ void cvt_f32_to_bf16(const float* __restrict__ in,
                                bf16_t* __restrict__ out, int n) {
  int i = blockIdx.x * blockDim.x + threadIdx.x;
  int stride = gridDim.x * blockDim.x;
  for (; i < n; i += stride) out[i] = (bf16_t)in[i];
}

// ---------------------------------------------------------------------------
// QKV projection GEMM:  Y[m][n] = sum_k X[m][k] * W[n][k]
// Block = 128 threads = 4 waves (2x2); block tile 128x128; wave tile 64x64.
// A (x) tile 128x32 staged in LDS by the TDM, double-buffered; B (weights)
// read directly from global (L2-resident).  16 WMMAs per wave per K-step.
// grid = (N/128, M/128, 3); z selects {Q,K,V}.  Output -> [B*H, T, HD] bf16.
// ---------------------------------------------------------------------------
__global__ __launch_bounds__(128)
void qkv_gemm(const bf16_t* __restrict__ x,
              const bf16_t* __restrict__ wq, const bf16_t* __restrict__ wk,
              const bf16_t* __restrict__ wv,
              bf16_t* __restrict__ qo, bf16_t* __restrict__ ko,
              bf16_t* __restrict__ vo) {
  __shared__ bf16_t xtile[2][128 * 32];   // 2 x 8 KB double buffer

  const bf16_t* w;
  bf16_t* y;
  if (blockIdx.z == 0)      { w = wq; y = qo; }
  else if (blockIdx.z == 1) { w = wk; y = ko; }
  else                      { w = wv; y = vo; }

  int wave = threadIdx.x >> 5;
  int wm = wave & 1, wn = wave >> 1;
  int m0b = blockIdx.y * 128;
  int m0 = m0b + wm * 64;
  int n0 = blockIdx.x * 128 + wn * 64;

  v8f acc[4][4];
#pragma unroll
  for (int i = 0; i < 4; ++i)
#pragma unroll
    for (int j = 0; j < 4; ++j)
#pragma unroll
      for (int e = 0; e < 8; ++e) acc[i][j][e] = 0.0f;

  const int NSTEP = D_SZ / 32;   // 32
  if (wave == 0) {
    tdm_load_2d_bf16((unsigned)(uintptr_t)(void*)&xtile[0][0],
                     x + (size_t)m0b * D_SZ, D_SZ, M_TOT, 32, 128, D_SZ);
  }

  for (int i = 0; i < NSTEP; ++i) {
    int kb = i * 32;
    if (wave == 0) {
      if (i + 1 < NSTEP) {
        tdm_load_2d_bf16((unsigned)(uintptr_t)(void*)&xtile[(i + 1) & 1][0],
                         x + (size_t)m0b * D_SZ + (kb + 32),
                         D_SZ, M_TOT, 32, 128, D_SZ);
        __builtin_amdgcn_s_wait_tensorcnt(1);   // buffer i landed
      } else {
        __builtin_amdgcn_s_wait_tensorcnt(0);
      }
    }
    __syncthreads();

    const bf16_t* xa = &xtile[i & 1][0];
    v16bf b[4];
#pragma unroll
    for (int nt = 0; nt < 4; ++nt)
      b[nt] = load_b_rowmajor(w + (size_t)(n0 + nt * 16) * D_SZ + kb, D_SZ);
#pragma unroll
    for (int mt = 0; mt < 4; ++mt) {
      v16bf a = load_a_frag(xa + (wm * 64 + mt * 16) * 32, 32);
#pragma unroll
      for (int nt = 0; nt < 4; ++nt)
        acc[mt][nt] = __builtin_amdgcn_wmma_f32_16x16x32_bf16(
            false, a, false, b[nt], (short)0, acc[mt][nt], false, false);
    }
    __syncthreads();   // everyone done reading buf[i&1] before TDM reuses it
  }

  // C layout store: VGPR r -> (row = r + 8*(lane>>4), col = lane&15)
  int l = threadIdx.x & 31;
  int col = l & 15, rb = (l >> 4) * 8;
#pragma unroll
  for (int mt = 0; mt < 4; ++mt)
#pragma unroll
    for (int nt = 0; nt < 4; ++nt)
#pragma unroll
      for (int r = 0; r < 8; ++r) {
        int m = m0 + mt * 16 + rb + r;           // token index in [0, B*T)
        int n = n0 + nt * 16 + col;              // feature index in [0, 1024)
        int bb = m >> 11, t = m & (T_SZ - 1);
        int h = n >> 6, d = n & 63;
        y[(((size_t)bb * H_SZ + h) * T_SZ + t) * HD_SZ + d] =
            (bf16_t)acc[mt][nt][r];
      }
}

// ---------------------------------------------------------------------------
// Flash attention (causal).  grid = (B*H, T/64), block = 128 threads (4 waves)
// Wave handles 16 query rows; per 64-key tile: S = Q K^T (8 WMMA), online
// softmax in C-layout registers, P -> LDS (bf16) -> A frags, O += P V (8 WMMA)
// ---------------------------------------------------------------------------
__global__ __launch_bounds__(128)
void flash_attn(const bf16_t* __restrict__ q, const bf16_t* __restrict__ k,
                const bf16_t* __restrict__ v, bf16_t* __restrict__ ctx) {
  __shared__ bf16_t lds_p[4 * 16 * 64];   // 8 KB, one 16x64 tile per wave

  int bh = blockIdx.x;                    // 0..63
  int qt = blockIdx.y;                    // 0..31 (query tile of 64)
  int wave = threadIdx.x >> 5;
  int l = threadIdx.x & 31;
  int col = l & 15, rb = (l >> 4) * 8;

  const size_t base = (size_t)bh * T_SZ * HD_SZ;
  int qrow0 = qt * 64 + wave * 16;

  v16bf qf0 = load_a_frag(q + base + (size_t)qrow0 * HD_SZ, HD_SZ);
  v16bf qf1 = load_a_frag(q + base + (size_t)qrow0 * HD_SZ + 32, HD_SZ);

  float m_i[8], l_i[8];
  v8f o[4];
#pragma unroll
  for (int r = 0; r < 8; ++r) { m_i[r] = -3.0e38f; l_i[r] = 0.0f; }
#pragma unroll
  for (int t = 0; t < 4; ++t)
#pragma unroll
    for (int e = 0; e < 8; ++e) o[t][e] = 0.0f;

  const float scale = 0.125f;  // 1/sqrt(64)
  bf16_t* myp = lds_p + wave * 16 * 64;

  for (int kb = 0; kb <= qt; ++kb) {
    // ---- S = Q K^T for 4 key subtiles of 16 ----
    v8f s[4];
#pragma unroll
    for (int t = 0; t < 4; ++t) {
      const bf16_t* kbase = k + base + (size_t)(kb * 64 + t * 16) * HD_SZ;
      v16bf kf0 = load_b_rowmajor(kbase, HD_SZ);
      v16bf kf1 = load_b_rowmajor(kbase + 32, HD_SZ);
      v8f c = {};
      c = __builtin_amdgcn_wmma_f32_16x16x32_bf16(false, qf0, false, kf0,
                                                  (short)0, c, false, false);
      c = __builtin_amdgcn_wmma_f32_16x16x32_bf16(false, qf1, false, kf1,
                                                  (short)0, c, false, false);
      s[t] = c;
    }

    // ---- scale + causal mask (element (qrow, key) in C layout) ----
#pragma unroll
    for (int t = 0; t < 4; ++t)
#pragma unroll
      for (int r = 0; r < 8; ++r) {
        int qrow = qrow0 + rb + r;
        int key = kb * 64 + t * 16 + col;
        float val = s[t][r] * scale;
        s[t][r] = (key <= qrow) ? val : -3.0e38f;
      }

    // ---- online softmax: rows live in 16-lane halves ----
#pragma unroll
    for (int r = 0; r < 8; ++r) {
      float mx = s[0][r];
#pragma unroll
      for (int t = 1; t < 4; ++t) mx = fmaxf(mx, s[t][r]);
      mx = fmaxf(mx, __shfl_xor(mx, 1, 32));
      mx = fmaxf(mx, __shfl_xor(mx, 2, 32));
      mx = fmaxf(mx, __shfl_xor(mx, 4, 32));
      mx = fmaxf(mx, __shfl_xor(mx, 8, 32));
      float m_new = fmaxf(m_i[r], mx);
      float alpha = __expf(m_i[r] - m_new);
      m_i[r] = m_new;
      float rs = 0.0f;
#pragma unroll
      for (int t = 0; t < 4; ++t) {
        float p = __expf(s[t][r] - m_new);
        s[t][r] = p;
        rs += p;
      }
      rs += __shfl_xor(rs, 1, 32);
      rs += __shfl_xor(rs, 2, 32);
      rs += __shfl_xor(rs, 4, 32);
      rs += __shfl_xor(rs, 8, 32);
      l_i[r] = l_i[r] * alpha + rs;
#pragma unroll
      for (int t = 0; t < 4; ++t) o[t][r] *= alpha;
    }

    // ---- P -> LDS (bf16 16x64 row-major, per-wave region) ----
#pragma unroll
    for (int t = 0; t < 4; ++t)
#pragma unroll
      for (int r = 0; r < 8; ++r)
        myp[(rb + r) * 64 + t * 16 + col] = (bf16_t)s[t][r];

    __syncthreads();   // uniform trip count across block

    // ---- O += P V ----
#pragma unroll
    for (int c2 = 0; c2 < 2; ++c2) {
      v16bf pf = load_a_frag(myp + c2 * 32, 64);
#pragma unroll
      for (int t = 0; t < 4; ++t) {
        v16bf vf = load_b_colstride(
            v + base + (size_t)(kb * 64 + c2 * 32) * HD_SZ + t * 16, HD_SZ);
        o[t] = __builtin_amdgcn_wmma_f32_16x16x32_bf16(
            false, pf, false, vf, (short)0, o[t], false, false);
      }
    }
    __syncthreads();
  }

  // ---- normalize, store ctx as bf16 [B, T, H*HD] ----
  int bb = bh >> 4, h = bh & 15;
#pragma unroll
  for (int t = 0; t < 4; ++t)
#pragma unroll
    for (int r = 0; r < 8; ++r) {
      int qrow = qrow0 + rb + r;
      float val = o[t][r] / l_i[r];
      ctx[((size_t)bb * T_SZ + qrow) * D_SZ + h * HD_SZ + t * 16 + col] =
          (bf16_t)val;
    }
}

// ---------------------------------------------------------------------------
// Output projection: out[m][n] = sum_k ctx[m][k]*Wo[n][k] + bias[n]  (f32 out)
// Same TDM-fed 64x64-per-wave structure as qkv_gemm.
// ---------------------------------------------------------------------------
__global__ __launch_bounds__(128)
void out_gemm(const bf16_t* __restrict__ ctx, const bf16_t* __restrict__ wo,
              const float* __restrict__ bias, float* __restrict__ out) {
  __shared__ bf16_t ctile[2][128 * 32];

  int wave = threadIdx.x >> 5;
  int wm = wave & 1, wn = wave >> 1;
  int m0b = blockIdx.y * 128;
  int m0 = m0b + wm * 64;
  int n0 = blockIdx.x * 128 + wn * 64;

  v8f acc[4][4];
#pragma unroll
  for (int i = 0; i < 4; ++i)
#pragma unroll
    for (int j = 0; j < 4; ++j)
#pragma unroll
      for (int e = 0; e < 8; ++e) acc[i][j][e] = 0.0f;

  const int NSTEP = D_SZ / 32;
  if (wave == 0) {
    tdm_load_2d_bf16((unsigned)(uintptr_t)(void*)&ctile[0][0],
                     ctx + (size_t)m0b * D_SZ, D_SZ, M_TOT, 32, 128, D_SZ);
  }

  for (int i = 0; i < NSTEP; ++i) {
    int kb = i * 32;
    if (wave == 0) {
      if (i + 1 < NSTEP) {
        tdm_load_2d_bf16((unsigned)(uintptr_t)(void*)&ctile[(i + 1) & 1][0],
                         ctx + (size_t)m0b * D_SZ + (kb + 32),
                         D_SZ, M_TOT, 32, 128, D_SZ);
        __builtin_amdgcn_s_wait_tensorcnt(1);
      } else {
        __builtin_amdgcn_s_wait_tensorcnt(0);
      }
    }
    __syncthreads();

    const bf16_t* ca = &ctile[i & 1][0];
    v16bf b[4];
#pragma unroll
    for (int nt = 0; nt < 4; ++nt)
      b[nt] = load_b_rowmajor(wo + (size_t)(n0 + nt * 16) * D_SZ + kb, D_SZ);
#pragma unroll
    for (int mt = 0; mt < 4; ++mt) {
      v16bf a = load_a_frag(ca + (wm * 64 + mt * 16) * 32, 32);
#pragma unroll
      for (int nt = 0; nt < 4; ++nt)
        acc[mt][nt] = __builtin_amdgcn_wmma_f32_16x16x32_bf16(
            false, a, false, b[nt], (short)0, acc[mt][nt], false, false);
    }
    __syncthreads();
  }

  int l = threadIdx.x & 31;
  int col = l & 15, rb = (l >> 4) * 8;
#pragma unroll
  for (int mt = 0; mt < 4; ++mt)
#pragma unroll
    for (int nt = 0; nt < 4; ++nt)
#pragma unroll
      for (int r = 0; r < 8; ++r) {
        int m = m0 + mt * 16 + rb + r;
        int n = n0 + nt * 16 + col;
        out[(size_t)m * D_SZ + n] = acc[mt][nt][r] + bias[n];
      }
}

// ---------------------------------------------------------------------------
// Host launcher
// ---------------------------------------------------------------------------
extern "C" void kernel_launch(void* const* d_in, const int* in_sizes, int n_in,
                              void* d_out, int out_size, void* d_ws,
                              size_t ws_size, hipStream_t stream) {
  (void)in_sizes; (void)n_in; (void)out_size; (void)ws_size;

  const float* x_f  = (const float*)d_in[0];
  const float* wq_f = (const float*)d_in[1];
  const float* wk_f = (const float*)d_in[2];
  const float* wv_f = (const float*)d_in[3];
  const float* wo_f = (const float*)d_in[4];
  const float* b_o  = (const float*)d_in[5];
  float* out = (float*)d_out;

  char* ws = (char*)d_ws;
  size_t off = 0;
  bf16_t* x_b  = (bf16_t*)(ws + off); off += (size_t)M_TOT * D_SZ * 2;   // 16 MB
  bf16_t* wq_b = (bf16_t*)(ws + off); off += (size_t)D_SZ * D_SZ * 2;    //  2 MB
  bf16_t* wk_b = (bf16_t*)(ws + off); off += (size_t)D_SZ * D_SZ * 2;
  bf16_t* wv_b = (bf16_t*)(ws + off); off += (size_t)D_SZ * D_SZ * 2;
  bf16_t* wo_b = (bf16_t*)(ws + off); off += (size_t)D_SZ * D_SZ * 2;
  bf16_t* q_b  = (bf16_t*)(ws + off); off += (size_t)M_TOT * D_SZ * 2;   // 16 MB
  bf16_t* k_b  = (bf16_t*)(ws + off); off += (size_t)M_TOT * D_SZ * 2;
  bf16_t* v_b  = (bf16_t*)(ws + off); off += (size_t)M_TOT * D_SZ * 2;
  bf16_t* ctx_b = (bf16_t*)(ws + off); off += (size_t)M_TOT * D_SZ * 2;

  const int nx = M_TOT * D_SZ;      // 8388608
  const int nw = D_SZ * D_SZ;       // 1048576
  cvt_f32_to_bf16<<<4096, 256, 0, stream>>>(x_f,  x_b,  nx);
  cvt_f32_to_bf16<<<1024, 256, 0, stream>>>(wq_f, wq_b, nw);
  cvt_f32_to_bf16<<<1024, 256, 0, stream>>>(wk_f, wk_b, nw);
  cvt_f32_to_bf16<<<1024, 256, 0, stream>>>(wv_f, wv_b, nw);
  cvt_f32_to_bf16<<<1024, 256, 0, stream>>>(wo_f, wo_b, nw);

  dim3 gQKV(D_SZ / 128, M_TOT / 128, 3);      // (8, 64, 3)
  qkv_gemm<<<gQKV, 128, 0, stream>>>(x_b, wq_b, wk_b, wv_b, q_b, k_b, v_b);

  dim3 gAttn(B_SZ * H_SZ, T_SZ / 64);         // (64, 32)
  flash_attn<<<gAttn, 128, 0, stream>>>(q_b, k_b, v_b, ctx_b);

  dim3 gOut(D_SZ / 128, M_TOT / 128);         // (8, 64)
  out_gemm<<<gOut, 128, 0, stream>>>(ctx_b, wo_b, b_o, out);
}
